// STGNN_22892175687814
// MI455X (gfx1250) — matrix operand order repaired and checked by reading, not compiled, not run on hardware
//
#include <hip/hip_runtime.h>
#include <cstdint>
#include <cstddef>

// ---------------------------------------------------------------------------
// Types for CDNA5 WMMA (wave32)
// ---------------------------------------------------------------------------
typedef __bf16 bf16_t;
typedef __attribute__((ext_vector_type(16))) __bf16 bf16x16;
typedef __attribute__((ext_vector_type(8)))  __bf16 bf16x8;
typedef __attribute__((ext_vector_type(8)))  float  f32x8;

union bf16_join {                 // pair two 8-vectors into one 16-vector
  struct { bf16x8 lo, hi; } p;
  bf16x16 v;
};

static __device__ __forceinline__ bf16_t f32_to_bf16(float f) {
  union { float f; unsigned u; } v; v.f = f;
  unsigned r = v.u + 0x7FFFu + ((v.u >> 16) & 1u);   // round-to-nearest-even
  union { unsigned short us; bf16_t b; } o; o.us = (unsigned short)(r >> 16);
  return o.b;
}
static __device__ __forceinline__ bf16_t bf16_zero() {
  union { unsigned short us; bf16_t b; } o; o.us = 0; return o.b;
}

// ---------------------------------------------------------------------------
// Dense GEMM: C[M,N] = A[M,Kp] @ B[Kp,N] (+bias, optional relu)
//   A  : bf16 [M x Kp] row-major, Kp % 32 == 0, pad cols zero
//   Bt : bf16 [N x Kp] = B transposed, pad K zero  -> contiguous lane frags
// Each wave computes a 16x64 C strip (4 WMMA tiles), reusing its A fragment.
// K loop is software-pipelined (double-buffered fragments) so WMMAs overlap
// the next k-step's global loads. Buffers carry 512B slack for the overshoot
// prefetch of the final iteration.
// ---------------------------------------------------------------------------
#define GEMM_WAVES 8
#define GEMM_NT    4    // N-tiles per wave (64 columns)

__global__ __launch_bounds__(256) void gemm_bf16_wmma_kernel(
    const bf16_t* __restrict__ A, const bf16_t* __restrict__ Bt,
    const float* __restrict__ bias, float* __restrict__ Cf,
    bf16_t* __restrict__ Cbf, int M, int N, int Kp, int doRelu)
{
  const int wave = threadIdx.x >> 5;
  const int lane = threadIdx.x & 31;
  const int tilesN4 = (N + 63) >> 6;
  const int tile = blockIdx.x * GEMM_WAVES + wave;
  if (tile >= (M >> 4) * tilesN4) return;

  const int tm   = tile / tilesN4;
  const int tn4  = tile - tm * tilesN4;
  const int half = lane >> 4;      // 0: lanes 0-15, 1: lanes 16-31
  const int l16  = lane & 15;
  const int row  = (tm << 4) + l16;

  const bf16_t* aRow = A + (size_t)row * (size_t)Kp;

  int  colj[GEMM_NT];
  bool okj[GEMM_NT];
  const bf16_t* bCol[GEMM_NT];
#pragma unroll
  for (int j = 0; j < GEMM_NT; ++j) {
    int c   = (tn4 << 6) + (j << 4) + l16;
    okj[j]  = c < N;
    colj[j] = c;
    // lane's B fragment: Bt[col][k + half*16 .. +15]  (32B aligned)
    bCol[j] = Bt + (size_t)(okj[j] ? c : (N - 1)) * (size_t)Kp + half * 16;
  }

  f32x8 acc[GEMM_NT] = {};

  // ---- pipeline prologue: fragments for k = 0
  bf16_join aC;
  aC.p.lo = *(const bf16x8*)(aRow + half * 16);
  aC.p.hi = *(const bf16x8*)(aRow + 8 + half * 16);
  bf16x16 bC[GEMM_NT];
#pragma unroll
  for (int j = 0; j < GEMM_NT; ++j) bC[j] = *(const bf16x16*)(bCol[j]);

#pragma unroll 2
  for (int k = 0; k < Kp; k += 32) {
    const int kn = k + 32;            // last iteration overshoots into slack
    // issue next k-step loads first (distinct regs -> partial loadcnt waits)
    bf16_join aN;
    aN.p.lo = *(const bf16x8*)(aRow + kn + half * 16);
    aN.p.hi = *(const bf16x8*)(aRow + kn + 8 + half * 16);
    bf16x16 bN[GEMM_NT];
#pragma unroll
    for (int j = 0; j < GEMM_NT; ++j) bN[j] = *(const bf16x16*)(bCol[j] + kn);

    __builtin_prefetch(aRow + k + 256, 0, 3);        // global_prefetch_b8

    // compute on current fragments while next loads are in flight
#pragma unroll
    for (int j = 0; j < GEMM_NT; ++j)
      acc[j] = __builtin_amdgcn_wmma_f32_16x16x32_bf16(
          false, aC.v, false, bC[j], (short)0, acc[j], false, false);

    aC = aN;
#pragma unroll
    for (int j = 0; j < GEMM_NT; ++j) bC[j] = bN[j];
  }

  // Epilogue: C layout — VGPR v: lanes 0-15 hold M=v, lanes 16-31 M=v+8
  const size_t rbase = (size_t)((tm << 4) + half * 8);
#pragma unroll
  for (int j = 0; j < GEMM_NT; ++j) {
    if (!okj[j]) continue;
    const int col = colj[j];
    const float bv = bias ? bias[col] : 0.f;
    float*  cr = Cf + rbase * (size_t)N + col;
    bf16_t* cb = Cbf ? (Cbf + rbase * (size_t)N + col) : (bf16_t*)nullptr;
#pragma unroll
    for (int v = 0; v < 8; ++v) {
      float val = acc[j][v] + bv;
      if (doRelu) val = fmaxf(val, 0.f);
      cr[(size_t)v * (size_t)N] = val;
      if (cb) cb[(size_t)v * (size_t)N] = f32_to_bf16(val);
    }
  }
}

// ---------------------------------------------------------------------------
// SpMM: out[rows[e], :] += vals[e] * dense[cols[e], :]   (atomic f32)
// ---------------------------------------------------------------------------
__global__ void spmm_edges_kernel(const int* __restrict__ rows,
                                  const int* __restrict__ cols,
                                  const float* __restrict__ vals,
                                  const float* __restrict__ dense,
                                  float* __restrict__ out, int D)
{
  const int e = blockIdx.x;
  const int r = rows[e];
  const int c = cols[e];
  const float v = vals[e];
  const float* src = dense + (size_t)c * (size_t)D;
  float* dst = out + (size_t)r * (size_t)D;
  for (int d = threadIdx.x; d < D; d += blockDim.x)
    atomicAdd(dst + d, v * src[d]);
}

// ---------------------------------------------------------------------------
// Attention fuse: beta = softmax([h.a, t.a]); emb = b0*h + b1*t  (bf16 out)
// One wave per node; wave32 shuffle reduction.
// ---------------------------------------------------------------------------
__global__ __launch_bounds__(256) void attn_fuse_kernel(
    const float* __restrict__ h, const float* __restrict__ t,
    const float* __restrict__ a, bf16_t* __restrict__ emb_bf, int N, int D)
{
  const int wave = threadIdx.x >> 5;
  const int lane = threadIdx.x & 31;
  const int n = blockIdx.x * 8 + wave;
  if (n >= N) return;
  const float* hr = h + (size_t)n * (size_t)D;
  const float* tr = t + (size_t)n * (size_t)D;
  float s1 = 0.f, s2 = 0.f;
  for (int d = lane; d < D; d += 32) {
    float av = a[d];
    s1 += hr[d] * av;
    s2 += tr[d] * av;
  }
#pragma unroll
  for (int off = 16; off > 0; off >>= 1) {
    s1 += __shfl_xor(s1, off, 32);
    s2 += __shfl_xor(s2, off, 32);
  }
  const float m  = fmaxf(s1, s2);
  const float e1 = expf(s1 - m);
  const float e2 = expf(s2 - m);
  const float inv = 1.f / (e1 + e2);
  const float b1 = e1 * inv, b2 = e2 * inv;
  bf16_t* er = emb_bf + (size_t)n * (size_t)D;
  for (int d = lane; d < D; d += 32)
    er[d] = f32_to_bf16(b1 * hr[d] + b2 * tr[d]);
}

// ---------------------------------------------------------------------------
// Conversions / elementwise helpers
// ---------------------------------------------------------------------------
// Weight [K,N] f32 -> TRANSPOSED [N,Kp] bf16, zero pad K..Kp
__global__ void w_to_bf16_t_kernel(const float* __restrict__ W,
                                   bf16_t* __restrict__ Wt,
                                   int K, int N, int Kp)
{
  long long i = (long long)blockIdx.x * blockDim.x + threadIdx.x;
  if (i >= (long long)N * Kp) return;
  int c = (int)(i / Kp);                 // output row = B column
  int r = (int)(i - (long long)c * Kp);  // output col = K index
  Wt[i] = (r < K) ? f32_to_bf16(W[(size_t)r * (size_t)N + c]) : bf16_zero();
}

// Activation [M,K] f32 -> [M,Kp] bf16 with zero-filled pad cols (K..Kp)
__global__ void a_to_bf16_pad_kernel(const float* __restrict__ X,
                                     bf16_t* __restrict__ Xb,
                                     int M, int K, int Kp)
{
  long long i = (long long)blockIdx.x * blockDim.x + threadIdx.x;
  if (i >= (long long)M * Kp) return;
  int r = (int)(i / Kp);
  int c = (int)(i - (long long)r * Kp);
  Xb[i] = (c < K) ? f32_to_bf16(X[(size_t)r * (size_t)K + c]) : bf16_zero();
}

__global__ void fill_zero_kernel(float* __restrict__ p, long long n)
{
  long long i = (long long)blockIdx.x * blockDim.x + threadIdx.x;
  if (i < n) p[i] = 0.f;
}

__global__ void relu_kernel(float* __restrict__ p, long long n)
{
  long long i = (long long)blockIdx.x * blockDim.x + threadIdx.x;
  if (i < n) p[i] = fmaxf(p[i], 0.f);
}

__global__ void scale_add_kernel(const float* __restrict__ scale,
                                 const float* __restrict__ additive,
                                 float* __restrict__ o_scale,
                                 float* __restrict__ o_add, int G)
{
  int i = blockIdx.x * blockDim.x + threadIdx.x;
  if (i < G) {
    o_scale[i] = 1.f / (1.f + expf(-scale[i]));
    o_add[i]   = expf(additive[i]);
  }
}

// ---------------------------------------------------------------------------
// Host launcher
// ---------------------------------------------------------------------------
extern "C" void kernel_launch(void* const* d_in, const int* in_sizes, int n_in,
                              void* d_out, int out_size, void* d_ws, size_t ws_size,
                              hipStream_t stream)
{
  (void)n_in; (void)out_size; (void)ws_size;
  const int N_ = 40000, G = 3000, Gp = 3008, C = 20;
  const int E = in_sizes[1];

  // ---- inputs (setup_inputs dict order)
  const float* x       = (const float*)d_in[0];
  const int*   rows    = (const int*)  d_in[1];
  const int*   colsI   = (const int*)  d_in[2];
  const float* vals    = (const float*)d_in[3];
  const float* W1      = (const float*)d_in[4];
  const float* W2      = (const float*)d_in[5];
  const float* W3      = (const float*)d_in[6];
  const float* W4      = (const float*)d_in[7];
  const float* W5      = (const float*)d_in[8];
  const float* enc1_w  = (const float*)d_in[9];   const float* enc1_b = (const float*)d_in[10];
  const float* enc2_w  = (const float*)d_in[11];  const float* enc2_b = (const float*)d_in[12];
  const float* enc3_w  = (const float*)d_in[13];  const float* enc3_b = (const float*)d_in[14];
  const float* zl_w    = (const float*)d_in[15];  const float* zl_b   = (const float*)d_in[16];
  const float* dec1_w  = (const float*)d_in[17];  const float* dec1_b = (const float*)d_in[18];
  const float* dec2_w  = (const float*)d_in[19];  const float* dec2_b = (const float*)d_in[20];
  const float* dec3_w  = (const float*)d_in[21];  const float* dec3_b = (const float*)d_in[22];
  const float* xbar_w  = (const float*)d_in[23];  const float* xbar_b = (const float*)d_in[24];
  const float* a1      = (const float*)d_in[25];
  const float* a2      = (const float*)d_in[26];
  const float* a3      = (const float*)d_in[27];
  const float* a4      = (const float*)d_in[28];
  const float* scaleIn = (const float*)d_in[29];
  const float* addIn   = (const float*)d_in[30];

  // ---- outputs (concatenated tuple)
  float* out       = (float*)d_out;
  float* out_cls   = out;                                      // [N, C]
  float* out_xbar  = out + (size_t)N_ * C;                     // [N, G]
  float* out_scale = out_xbar + (size_t)N_ * G;                // [G]
  float* out_add   = out_scale + G;                            // [G]

  // ---- workspace layout (512B slack per region for pipeline overshoot)
  char* ws = (char*)d_ws;
  size_t off = 0;
  auto alloc = [&](size_t bytes) -> void* {
    void* p = ws + off;
    off = (off + bytes + 512 + 255) & ~(size_t)255;
    return p;
  };
  bf16_t* x_bf  = (bf16_t*)alloc((size_t)N_ * Gp * 2);
  float*  t1    = (float*) alloc((size_t)N_ * 512 * 4);
  float*  t2    = (float*) alloc((size_t)N_ * 256 * 4);
  float*  t3    = (float*) alloc((size_t)N_ * 128 * 4);
  float*  zbuf  = (float*) alloc((size_t)N_ * 128 * 4);
  float*  ftmp  = (float*) alloc((size_t)N_ * 512 * 4);
  float*  xw    = (float*) alloc((size_t)N_ * 512 * 4);
  float*  hbuf  = (float*) alloc((size_t)N_ * 512 * 4);
  bf16_t* bf0   = (bf16_t*)alloc((size_t)N_ * 512 * 2);
  bf16_t* bf1   = (bf16_t*)alloc((size_t)N_ * 512 * 2);
  // transposed bf16 weights: [N x Kp]
  bf16_t* W1b   = (bf16_t*)alloc((size_t)512 * Gp  * 2);
  bf16_t* W2b   = (bf16_t*)alloc((size_t)256 * 512 * 2);
  bf16_t* W3b   = (bf16_t*)alloc((size_t)128 * 256 * 2);
  bf16_t* W4b   = (bf16_t*)alloc((size_t)128 * 128 * 2);
  bf16_t* W5b   = (bf16_t*)alloc((size_t)C   * 128 * 2);
  bf16_t* enc1b = (bf16_t*)alloc((size_t)512 * Gp  * 2);
  bf16_t* enc2b = (bf16_t*)alloc((size_t)256 * 512 * 2);
  bf16_t* enc3b = (bf16_t*)alloc((size_t)128 * 256 * 2);
  bf16_t* zlb   = (bf16_t*)alloc((size_t)128 * 128 * 2);
  bf16_t* dec1b = (bf16_t*)alloc((size_t)128 * 128 * 2);
  bf16_t* dec2b = (bf16_t*)alloc((size_t)256 * 128 * 2);
  bf16_t* dec3b = (bf16_t*)alloc((size_t)512 * 256 * 2);
  bf16_t* xbarb = (bf16_t*)alloc((size_t)G   * 512 * 2);

  // ---- helpers
  auto ew_blocks = [](long long n) { return (unsigned)((n + 255) / 256); };

  auto convW = [&](const float* W, bf16_t* Wt, int K, int Nw, int Kp) {
    long long tot = (long long)Nw * Kp;
    hipLaunchKernelGGL(w_to_bf16_t_kernel, dim3(ew_blocks(tot)), dim3(256), 0, stream,
                       W, Wt, K, Nw, Kp);
  };
  auto gemm = [&](const bf16_t* A, const bf16_t* Bt, const float* bias,
                  float* Cf, bf16_t* Cbf, int M, int Nn, int Kp, int relu) {
    int tiles = (M / 16) * ((Nn + 63) / 64);
    int blocks = (tiles + GEMM_WAVES - 1) / GEMM_WAVES;
    hipLaunchKernelGGL(gemm_bf16_wmma_kernel, dim3(blocks), dim3(256), 0, stream,
                       A, Bt, bias, Cf, Cbf, M, Nn, Kp, relu);
  };
  auto zero = [&](float* p, long long n) {
    hipLaunchKernelGGL(fill_zero_kernel, dim3(ew_blocks(n)), dim3(256), 0, stream, p, n);
  };
  auto relu = [&](float* p, long long n) {
    hipLaunchKernelGGL(relu_kernel, dim3(ew_blocks(n)), dim3(256), 0, stream, p, n);
  };
  auto spmm = [&](const float* dense, float* o, int D) {
    hipLaunchKernelGGL(spmm_edges_kernel, dim3(E), dim3(64), 0, stream,
                       rows, colsI, vals, dense, o, D);
  };
  auto attn = [&](const float* h, const float* t, const float* a, bf16_t* e, int D) {
    hipLaunchKernelGGL(attn_fuse_kernel, dim3((N_ + 7) / 8), dim3(256), 0, stream,
                       h, t, a, e, N_, D);
  };

  // ---- stage bf16 operands
  {
    long long tot = (long long)N_ * Gp;
    hipLaunchKernelGGL(a_to_bf16_pad_kernel, dim3(ew_blocks(tot)), dim3(256), 0, stream,
                       x, x_bf, N_, G, Gp);
  }
  convW(W1,     W1b,   G,   512, Gp);
  convW(W2,     W2b,   512, 256, 512);
  convW(W3,     W3b,   256, 128, 256);
  convW(W4,     W4b,   128, 128, 128);
  convW(W5,     W5b,   128, C,   128);
  convW(enc1_w, enc1b, G,   512, Gp);
  convW(enc2_w, enc2b, 512, 256, 512);
  convW(enc3_w, enc3b, 256, 128, 256);
  convW(zl_w,   zlb,   128, 128, 128);
  convW(dec1_w, dec1b, 128, 128, 128);
  convW(dec2_w, dec2b, 128, 256, 128);
  convW(dec3_w, dec3b, 256, 512, 256);
  convW(xbar_w, xbarb, 512, G,   512);

  // ---- trivial outputs
  hipLaunchKernelGGL(scale_add_kernel, dim3((G + 255) / 256), dim3(256), 0, stream,
                     scaleIn, addIn, out_scale, out_add, G);

  // ---- autoencoder chain (bf16 A ping-pong between bf0/bf1)
  gemm(x_bf, enc1b, enc1_b, t1,       bf0,     N_, 512, Gp,  1);  // t1
  gemm(bf0,  enc2b, enc2_b, t2,       bf1,     N_, 256, 512, 1);  // t2
  gemm(bf1,  enc3b, enc3_b, t3,       bf0,     N_, 128, 256, 1);  // t3
  gemm(bf0,  zlb,   zl_b,   zbuf,     bf1,     N_, 128, 128, 0);  // z (no relu)
  gemm(bf1,  dec1b, dec1_b, ftmp,     bf0,     N_, 128, 128, 1);  // d1
  gemm(bf0,  dec2b, dec2_b, ftmp,     bf1,     N_, 256, 128, 1);  // d2
  gemm(bf1,  dec3b, dec3_b, ftmp,     bf0,     N_, 512, 256, 1);  // d3
  gemm(bf0,  xbarb, xbar_b, out_xbar, nullptr, N_, G,   512, 0);  // x_bar

  // ---- GCN stack with attention-fused skip connections
  // layer 1: h = relu(spmm(adj, x @ W1)); emb = attn(h, t1, a1)
  gemm(x_bf, W1b, nullptr, xw, nullptr, N_, 512, Gp, 0);
  zero(hbuf, (long long)N_ * 512);
  spmm(xw, hbuf, 512);
  relu(hbuf, (long long)N_ * 512);
  attn(hbuf, t1, a1, bf0, 512);

  // layer 2
  gemm(bf0, W2b, nullptr, xw, nullptr, N_, 256, 512, 0);
  zero(hbuf, (long long)N_ * 256);
  spmm(xw, hbuf, 256);
  relu(hbuf, (long long)N_ * 256);
  attn(hbuf, t2, a2, bf1, 256);

  // layer 3
  gemm(bf1, W3b, nullptr, xw, nullptr, N_, 128, 256, 0);
  zero(hbuf, (long long)N_ * 128);
  spmm(xw, hbuf, 128);
  relu(hbuf, (long long)N_ * 128);
  attn(hbuf, t3, a3, bf0, 128);

  // layer 4
  gemm(bf0, W4b, nullptr, xw, nullptr, N_, 128, 128, 0);
  zero(hbuf, (long long)N_ * 128);
  spmm(xw, hbuf, 128);
  relu(hbuf, (long long)N_ * 128);
  attn(hbuf, zbuf, a4, bf1, 128);

  // layer 5 (no relu): output = spmm(adj, emb @ W5)
  gemm(bf1, W5b, nullptr, xw, nullptr, N_, C, 128, 0);
  zero(out_cls, (long long)N_ * C);
  spmm(xw, out_cls, C);
}
